// ResidualMambaBlock_45037027066524
// MI455X (gfx1250) — matrix operand order repaired and checked by reading, not compiled, not run on hardware
//
#include <hip/hip_runtime.h>

// ---------------- problem constants (from reference) ----------------
#define Bc   2
#define Lc   2048
#define DMc  1024
#define DIc  2048          // 2*DM
#define DCc  4
#define DTc  64            // DM/16
#define Nst  16
#define Mrows (Bc*Lc)      // 4096
#define XZW  (2*DIc)       // 4096 columns of xz

typedef unsigned short u16;
typedef __bf16 bf16_t;
typedef bf16_t v16bf __attribute__((ext_vector_type(16)));
typedef bf16_t v8bf  __attribute__((ext_vector_type(8)));
typedef float  v8f   __attribute__((ext_vector_type(8)));
typedef unsigned int v4u __attribute__((ext_vector_type(4)));
typedef int          v4i __attribute__((ext_vector_type(4)));
typedef int          v8i __attribute__((ext_vector_type(8)));

union AFrag { v16bf v; v8bf h[2]; };
union BFrag { v16bf v; v8bf h[2]; };

__device__ __forceinline__ u16 f2bf(float f) {
    unsigned int u = __float_as_uint(f);
    u += 0x7fffu + ((u >> 16) & 1u);     // round-to-nearest-even
    return (u16)(u >> 16);
}
__device__ __forceinline__ float silu(float x) {
    return x / (1.f + __expf(-x));
}

// Issue a TDM copy of one contiguous chunk (multiple of 8 bytes) from
// global -> LDS.  Descriptor per CDNA5 ISA (D# groups 0/1),
// 1-D tile of 8-byte elements.  Tracked by TENSORcnt.
__device__ __forceinline__ void tdm_copy_to_lds(const void* gptr, unsigned ldsAddr,
                                                int chunk8 /* 8-byte units */)
{
    unsigned long long ga = (unsigned long long)(uintptr_t)gptr;
    v4u g0;
    g0.x = 1u;                                            // count=1, user descriptor
    g0.y = ldsAddr;                                       // lds_addr (bytes)
    g0.z = (unsigned)(ga & 0xffffffffu);                  // global_addr[31:0]
    g0.w = (unsigned)((ga >> 32) & 0x01ffffffu)           // global_addr[56:32]
         | 0x80000000u;                                   // type=2 ("image")
    v8i g1 = (v8i){};
    g1.s0 = (3 << 16);                                    // data_size = 8 bytes
    g1.s1 = (int)((chunk8 & 0xffff) << 16);               // tensor_dim0 lo16
    g1.s2 = (int)(((chunk8 >> 16) & 0xffff) | (1 << 16)); // tensor_dim0 hi | tensor_dim1=1
    g1.s3 = (int)((chunk8 & 0xffff) << 16);               // tile_dim0
    g1.s4 = 1;                                            // tile_dim1 = 1
    g1.s5 = chunk8;                                       // tensor_dim0_stride
    g1.s6 = 0; g1.s7 = 0;
    v4i z4 = (v4i){};
#if __clang_major__ >= 23
    v8i z8 = (v8i){};
    __builtin_amdgcn_tensor_load_to_lds(g0, g1, z4, z4, z8, 0);
#else
    __builtin_amdgcn_tensor_load_to_lds(g0, g1, z4, z4, 0);
#endif
}

// ---------------- weight swizzle: row-major f32 (K x N) -> WMMA B-frag order bf16 ----
__global__ __launch_bounds__(256) void swizzle_b_kernel(
    const float* __restrict__ W, u16* __restrict__ out, int K, int N)
{
    int tid = blockIdx.x * 256 + threadIdx.x;
    if (tid >= K * N) return;
    int e    = tid & 15;
    int l    = (tid >> 4) & 31;
    int tile = tid >> 9;
    int NT   = N >> 4;
    int kt   = tile / NT;
    int nt   = tile - kt * NT;
    int n    = nt * 16 + (l & 15);
    int k    = kt * 32 + ((e < 8) ? 0 : 16) + ((l < 16) ? 0 : 8) + (e & 7);
    out[tid] = f2bf(W[(size_t)k * N + n]);
}

// ---------------- LayerNorm (one block per row), output bf16 ----------------
__global__ __launch_bounds__(256) void ln_kernel(
    const float* __restrict__ x, const float* __restrict__ g,
    const float* __restrict__ bta, u16* __restrict__ xn)
{
    __shared__ float red0[8], red1[8];
    const int row = blockIdx.x;
    const float* xr = x + (size_t)row * DMc;
    float s = 0.f, s2 = 0.f;
    for (int i = threadIdx.x; i < DMc; i += 256) {
        float v = xr[i]; s += v; s2 += v * v;
    }
    for (int off = 16; off; off >>= 1) {
        s  += __shfl_down(s,  off, 32);
        s2 += __shfl_down(s2, off, 32);
    }
    int wv = threadIdx.x >> 5;
    if ((threadIdx.x & 31) == 0) { red0[wv] = s; red1[wv] = s2; }
    __syncthreads();
    if (threadIdx.x == 0) {
        float a = 0.f, b = 0.f;
        for (int i = 0; i < 8; i++) { a += red0[i]; b += red1[i]; }
        red0[0] = a; red1[0] = b;
    }
    __syncthreads();
    float mu  = red0[0] * (1.f / DMc);
    float var = red1[0] * (1.f / DMc) - mu * mu;
    float inv = rsqrtf(var + 1e-5f);
    for (int i = threadIdx.x; i < DMc; i += 256) {
        float v = (xr[i] - mu) * inv * g[i] + bta[i];
        xn[(size_t)row * DMc + i] = f2bf(v);
    }
}

// ---------------- WMMA bf16 GEMM with TDM-staged, double-buffered B tiles ----------
// C[M,N] = A[M,K] * B[K,N] (+ epilogue).  A row-major bf16; B pre-swizzled so each
// (kt, block) weight chunk is contiguous (NT_W*1024 bytes) -> single TDM descriptor.
// All NT_W B fragments are pulled from LDS into registers BEFORE the WMMA chain so
// the matrix pipe issues back-to-back (no per-WMMA s_wait_dscnt 0).
// EPI: 0 plain f32 | 1 f32 + bf16(dt cols) | 2 softplus(acc+bias) | 3 acc+addend
template<int NT_W, int EPI>
__global__ __launch_bounds__(256) void wmma_gemm_kernel(
    const u16* __restrict__ A, const u16* __restrict__ Bswz,
    float* __restrict__ C, int M, int N, int K,
    const float* __restrict__ bias, const float* __restrict__ addend,
    u16* __restrict__ aux)
{
    __shared__ __align__(32) u16 bsh[2][NT_W * 512];

    const int lane = threadIdx.x & 31;
    const int wave = threadIdx.x >> 5;
    const int m0   = blockIdx.y * (8 * 16) + wave * 16;
    const int n0   = blockIdx.x * (NT_W * 16);
    const int NT   = N >> 4;
    const int KT   = K >> 5;
    const int chunk8 = NT_W * 128;                     // chunk size in 8-byte units

    const u16* bchunk0 = Bswz + (size_t)(n0 >> 4) * 512;   // kt stride = NT*512 elems
    const size_t ktStride = (size_t)NT * 512;
    unsigned ldsA[2] = { (unsigned)(uintptr_t)&bsh[0][0],
                         (unsigned)(uintptr_t)&bsh[1][0] };

    v8f acc[NT_W];
    #pragma unroll
    for (int i = 0; i < NT_W; i++) acc[i] = (v8f){};

    // prologue: TDM-stage chunk 0 into buffer 0
    if (wave == 0) {
        tdm_copy_to_lds(bchunk0, ldsA[0], chunk8);
        __builtin_amdgcn_s_wait_tensorcnt((short)0);
    }
    __syncthreads();

    const int mrow   = m0 + (lane & 15);
    const int a_koff = (lane < 16) ? 0 : 8;            // 16-bit A 16x32 lane layout
    const u16* Arow  = A + (size_t)mrow * K;

    AFrag a;
    a.h[0] = *(const v8bf*)(Arow + a_koff);
    a.h[1] = *(const v8bf*)(Arow + 16 + a_koff);

    for (int kt = 0; kt < KT; ++kt) {
        const bool more = (kt + 1) < KT;
        // overlap: TDM for next chunk into the other buffer
        if (wave == 0 && more)
            tdm_copy_to_lds(bchunk0 + (size_t)(kt + 1) * ktStride,
                            ldsA[(kt + 1) & 1], chunk8);
        // overlap: next A fragment from global into registers
        AFrag an;
        if (more) {
            const u16* Ak = Arow + (kt + 1) * 32;
            an.h[0] = *(const v8bf*)(Ak + a_koff);
            an.h[1] = *(const v8bf*)(Ak + 16 + a_koff);
        }
        // pull ALL B fragments for this k-step out of LDS first ...
        const u16* bl = &bsh[kt & 1][lane * 16];
        BFrag b[NT_W];
        #pragma unroll
        for (int nt = 0; nt < NT_W; nt++)
            b[nt].v = *(const v16bf*)(bl + nt * 512);
        // ... then run the WMMA chain back-to-back on the matrix pipe
        #pragma unroll
        for (int nt = 0; nt < NT_W; nt++)
            acc[nt] = __builtin_amdgcn_wmma_f32_16x16x32_bf16(
                false, a.v, false, b[nt].v, (short)0, acc[nt], false, false);

        if (wave == 0 && more)
            __builtin_amdgcn_s_wait_tensorcnt((short)0);
        __syncthreads();
        a = an;
    }

    // C layout: VGPR r -> row m0 + r (+8 for lanes 16..31), col n0 + (lane&15)
    const int colbase = n0 + (lane & 15);
    const int rbase   = (lane < 16) ? 0 : 8;
    #pragma unroll
    for (int nt = 0; nt < NT_W; nt++) {
        int col = colbase + nt * 16;
        #pragma unroll
        for (int r = 0; r < 8; r++) {
            int row = m0 + rbase + r;
            float v = acc[nt][r];
            size_t o = (size_t)row * N + col;
            if (EPI == 0) {
                C[o] = v;
            } else if (EPI == 1) {
                C[o] = v;
                if (col < DTc) aux[(size_t)row * DTc + col] = f2bf(v);
            } else if (EPI == 2) {
                float t = v + bias[col];
                C[o] = (t > 20.f) ? t : log1pf(__expf(t));   // softplus
            } else {
                C[o] = v + addend[o];
            }
        }
    }
}

// ---------------- depthwise causal conv (DC=4) + bias + SiLU -> u (f32 + bf16) --------
__global__ __launch_bounds__(256) void conv_silu_kernel(
    const float* __restrict__ xz, const float* __restrict__ cw,
    const float* __restrict__ cb, float* __restrict__ u, u16* __restrict__ ub)
{
    int tid = blockIdx.x * 256 + threadIdx.x;      // B*L*DI threads
    int d = tid & (DIc - 1);
    int l = (tid >> 11) & (Lc - 1);
    int b = tid >> 22;
    size_t base = (size_t)(b * Lc) * XZW;          // xi = cols [0, DI) of xz
    float acc = cb[d];
    #pragma unroll
    for (int j = 0; j < DCc; j++) {
        int t = l - (DCc - 1) + j;
        if (t >= 0) acc += xz[base + (size_t)t * XZW + d] * cw[d * DCc + j];
    }
    float s = silu(acc);
    size_t o = ((size_t)(b * Lc + l)) * DIc + d;
    u[o] = s;
    ub[o] = f2bf(s);
}

// ---------------- selective scan: lane = (b,d,n); 16-lane shfl_xor reduce for y -------
__global__ __launch_bounds__(256) void scan_kernel(
    const float* __restrict__ delta, const float* __restrict__ u,
    const float* __restrict__ dbl, const float* __restrict__ A_log,
    const float* __restrict__ Dskip, float* __restrict__ ys)
{
    int gid = blockIdx.x * 256 + threadIdx.x;      // B*DI*N = 65536 threads
    int n = gid & (Nst - 1);
    int d = (gid >> 4) & (DIc - 1);
    int b = gid >> 15;
    float A   = -__expf(A_log[d * Nst + n]);
    float Dsk = Dskip[d];
    float h = 0.f;
    for (int t = 0; t < Lc; t++) {
        size_t r  = (size_t)(b * Lc + t);
        float dl  = delta[r * DIc + d];
        float uu  = u[r * DIc + d];
        float Bm  = dbl[r * 96 + DTc + n];
        float Cm  = dbl[r * 96 + DTc + Nst + n];
        h = __expf(dl * A) * h + dl * uu * Bm;
        float y = h * Cm;
        y += __shfl_xor(y, 1);
        y += __shfl_xor(y, 2);
        y += __shfl_xor(y, 4);
        y += __shfl_xor(y, 8);
        if (n == 0) ys[r * DIc + d] = y + uu * Dsk;
    }
}

// ---------------- gate: yb = bf16( ys * silu(z) ) ----------------
__global__ __launch_bounds__(256) void gate_kernel(
    const float* __restrict__ ys, const float* __restrict__ xz, u16* __restrict__ yb)
{
    int tid = blockIdx.x * 256 + threadIdx.x;      // B*L*DI threads
    int d = tid & (DIc - 1);
    int r = tid >> 11;                             // row = b*L + l
    float z = xz[(size_t)r * XZW + DIc + d];
    yb[(size_t)r * DIc + d] = f2bf(ys[(size_t)r * DIc + d] * silu(z));
}

// ---------------- host-side orchestration ----------------
extern "C" void kernel_launch(void* const* d_in, const int* in_sizes, int n_in,
                              void* d_out, int out_size, void* d_ws, size_t ws_size,
                              hipStream_t stream)
{
    const float* x      = (const float*)d_in[0];
    const float* ln_g   = (const float*)d_in[1];
    const float* ln_b   = (const float*)d_in[2];
    const float* W_in   = (const float*)d_in[3];
    const float* conv_w = (const float*)d_in[4];
    const float* conv_b = (const float*)d_in[5];
    const float* W_x    = (const float*)d_in[6];
    const float* W_dt   = (const float*)d_in[7];
    const float* b_dt   = (const float*)d_in[8];
    const float* A_log  = (const float*)d_in[9];
    const float* Dskip  = (const float*)d_in[10];
    const float* W_out  = (const float*)d_in[11];
    float* out = (float*)d_out;

    char* ws = (char*)d_ws;
    size_t off = 0;
    auto alloc = [&](size_t bytes) -> char* {
        off = (off + 255) & ~(size_t)255;
        char* p = ws + off;
        off += bytes;
        return p;
    };

    u16*   xn16  = (u16*)  alloc((size_t)Mrows * DMc * 2);
    u16*   WinS  = (u16*)  alloc((size_t)DMc * XZW * 2);
    u16*   WxS   = (u16*)  alloc((size_t)DIc * 96 * 2);
    u16*   WdtS  = (u16*)  alloc((size_t)DTc * DIc * 2);
    u16*   WoutS = (u16*)  alloc((size_t)DIc * DMc * 2);
    float* xz    = (float*)alloc((size_t)Mrows * XZW * 4);
    float* u     = (float*)alloc((size_t)Mrows * DIc * 4);
    u16*   ub    = (u16*)  alloc((size_t)Mrows * DIc * 2);
    float* dbl   = (float*)alloc((size_t)Mrows * 96 * 4);
    u16*   dt16  = (u16*)  alloc((size_t)Mrows * DTc * 2);
    float* delta = (float*)alloc((size_t)Mrows * DIc * 4);
    float* ys    = (float*)alloc((size_t)Mrows * DIc * 4);
    u16*   yb    = (u16*)  alloc((size_t)Mrows * DIc * 2);

    // 1) weight swizzles (bf16 fragment order)
    swizzle_b_kernel<<<(DMc * XZW + 255) / 256, 256, 0, stream>>>(W_in,  WinS,  DMc, XZW);
    swizzle_b_kernel<<<(DIc * 96  + 255) / 256, 256, 0, stream>>>(W_x,   WxS,   DIc, 96);
    swizzle_b_kernel<<<(DTc * DIc + 255) / 256, 256, 0, stream>>>(W_dt,  WdtS,  DTc, DIc);
    swizzle_b_kernel<<<(DIc * DMc + 255) / 256, 256, 0, stream>>>(W_out, WoutS, DIc, DMc);

    // 2) LayerNorm -> bf16
    ln_kernel<<<Mrows, 256, 0, stream>>>(x, ln_g, ln_b, xn16);

    // 3) xz = xn @ W_in   (4096 x 1024 x 4096)
    wmma_gemm_kernel<8, 0><<<dim3(XZW / 128, Mrows / 128), 256, 0, stream>>>(
        xn16, WinS, xz, Mrows, XZW, DMc, nullptr, nullptr, nullptr);

    // 4) depthwise conv + SiLU -> u
    conv_silu_kernel<<<(Bc * Lc * DIc) / 256, 256, 0, stream>>>(xz, conv_w, conv_b, u, ub);

    // 5) dbl = u @ W_x   (4096 x 2048 x 96), also emit dt bf16
    wmma_gemm_kernel<6, 1><<<dim3(1, Mrows / 128), 256, 0, stream>>>(
        ub, WxS, dbl, Mrows, 96, DIc, nullptr, nullptr, dt16);

    // 6) delta = softplus(dt @ W_dt + b_dt)   (4096 x 64 x 2048)
    wmma_gemm_kernel<8, 2><<<dim3(DIc / 128, Mrows / 128), 256, 0, stream>>>(
        dt16, WdtS, delta, Mrows, DIc, DTc, b_dt, nullptr, nullptr);

    // 7) selective scan (+ u*Dskip)
    scan_kernel<<<(Bc * DIc * Nst) / 256, 256, 0, stream>>>(delta, u, dbl, A_log, Dskip, ys);

    // 8) gate with silu(z) -> bf16
    gate_kernel<<<(Bc * Lc * DIc) / 256, 256, 0, stream>>>(ys, xz, yb);

    // 9) out = x + y @ W_out   (4096 x 2048 x 1024)
    wmma_gemm_kernel<8, 3><<<dim3(DMc / 128, Mrows / 128), 256, 0, stream>>>(
        yb, WoutS, out, Mrows, DMc, DIc, nullptr, x, nullptr);

    (void)in_sizes; (void)n_in; (void)out_size; (void)ws_size;
}
